// QuantLinear_7687991459869
// MI455X (gfx1250) — 8-bit matrix operand order repaired and checked by reading, compiled, not run on hardware
//
#include <hip/hip_runtime.h>
#include <stdint.h>


typedef int v8i __attribute__((ext_vector_type(8)));

#define DIN   512
#define DOUT  512
#define MTOT  (8 * 8192)

static constexpr size_t OFF_XQ = 256;
static constexpr size_t OFF_WQ = OFF_XQ + (size_t)MTOT * DIN;
static constexpr size_t OFF_S  = OFF_WQ + (size_t)DOUT * DIN;
static constexpr size_t OFF_B  = OFF_S  + (size_t)DOUT * 4;

__global__ void k_init(unsigned* mx) {
    if (threadIdx.x == 0 && blockIdx.x == 0) *mx = 0u;
}

__global__ void k_absmax(const float4* __restrict__ x, unsigned* __restrict__ mx, int n4) {
    unsigned local = 0u;
    int stride = gridDim.x * blockDim.x;
    for (int i = blockIdx.x * blockDim.x + threadIdx.x; i < n4; i += stride) {
        float4 v = x[i];
        local = max(local, __float_as_uint(fabsf(v.x)));
        local = max(local, __float_as_uint(fabsf(v.y)));
        local = max(local, __float_as_uint(fabsf(v.z)));
        local = max(local, __float_as_uint(fabsf(v.w)));
    }
    __shared__ unsigned red[256];
    red[threadIdx.x] = local;
    __syncthreads();
    for (int s = 128; s > 0; s >>= 1) {
        if (threadIdx.x < s)
            red[threadIdx.x] = max(red[threadIdx.x], red[threadIdx.x + s]);
        __syncthreads();
    }
    if (threadIdx.x == 0) atomicMax(mx, red[0]);
}

__global__ void k_quant_w(const float* __restrict__ w, const float* __restrict__ bias,
                          int8_t* __restrict__ wq, float* __restrict__ s,
                          float* __restrict__ bq, const unsigned* __restrict__ mx) {
    int row = blockIdx.x;
    int tid = threadIdx.x;
    const float* wr = w + (size_t)row * DIN;
    float v0 = wr[tid];
    float v1 = wr[tid + 256];

    __shared__ float red[256];
    red[tid] = fmaxf(fabsf(v0), fabsf(v1));
    __syncthreads();
    for (int st = 128; st > 0; st >>= 1) {
        if (tid < st) red[tid] = fmaxf(red[tid], red[tid + st]);
        __syncthreads();
    }
    __shared__ float wsc_sh;
    if (tid == 0) wsc_sh = red[0] / 127.0f;
    __syncthreads();
    float wsc = wsc_sh;
    float inv = (wsc > 0.0f) ? (1.0f / wsc) : 0.0f;

    int q0 = (int)rintf(v0 * inv);  q0 = min(127, max(-128, q0));
    int q1 = (int)rintf(v1 * inv);  q1 = min(127, max(-128, q1));
    wq[(size_t)row * DIN + tid]       = (int8_t)q0;
    wq[(size_t)row * DIN + tid + 256] = (int8_t)q1;

    if (tid == 0) {
        float act = __uint_as_float(*mx) / 127.0f;
        float comb = act * wsc;
        s[row] = comb;
        float t = rintf(bias[row] / comb);
        t = fminf(fmaxf(t, -2147483648.0f), 2147483647.0f);
        bq[row] = t * comb;
    }
}

__global__ void k_quant_x(const float4* __restrict__ x, int* __restrict__ xq,
                          const unsigned* __restrict__ mx, int n4) {
    float act = __uint_as_float(*mx) / 127.0f;
    float inv = (act > 0.0f) ? (1.0f / act) : 0.0f;
    int stride = gridDim.x * blockDim.x;
    for (int i = blockIdx.x * blockDim.x + threadIdx.x; i < n4; i += stride) {
        float4 v = x[i];
        int q0 = min(127, max(-128, (int)rintf(v.x * inv)));
        int q1 = min(127, max(-128, (int)rintf(v.y * inv)));
        int q2 = min(127, max(-128, (int)rintf(v.z * inv)));
        int q3 = min(127, max(-128, (int)rintf(v.w * inv)));
        xq[i] = (q0 & 0xFF) | ((q1 & 0xFF) << 8) | ((q2 & 0xFF) << 16) | ((q3 & 0xFF) << 24);
    }
}

__global__ __launch_bounds__(256, 1)
void k_gemm_iu8(const int8_t* __restrict__ xq, const int8_t* __restrict__ wq,
                const float* __restrict__ s, const float* __restrict__ bq,
                float* __restrict__ out) {
    const int lane  = threadIdx.x & 31;
    const int wave  = threadIdx.x >> 5;
    const int waveM = wave & 3;
    const int waveN = wave >> 2;
    const int half  = lane >> 4;
    const int l15   = lane & 15;

    const int m_base = blockIdx.x * 128 + waveM * 32;
    const int n_base = blockIdx.y * 128 + waveN * 64;

    const v8i zero = {0, 0, 0, 0, 0, 0, 0, 0};
    v8i acc[2][4];
#pragma unroll
    for (int mi = 0; mi < 2; ++mi)
#pragma unroll
        for (int ni = 0; ni < 4; ++ni) acc[mi][ni] = zero;

    const int8_t* aptr[2];
#pragma unroll
    for (int mi = 0; mi < 2; ++mi)
        aptr[mi] = xq + (size_t)(m_base + mi * 16 + l15) * DIN + half * 16;
    const int8_t* bptr[4];
#pragma unroll
    for (int ni = 0; ni < 4; ++ni)
        bptr[ni] = wq + (size_t)(n_base + ni * 16 + l15) * DIN + half * 16;

#pragma unroll
    for (int k0 = 0; k0 < DIN; k0 += 64) {
        if (k0 + 64 < DIN) {
            __builtin_prefetch(aptr[0] + k0 + 64, 0, 3);
            __builtin_prefetch(aptr[1] + k0 + 64, 0, 3);
            __builtin_prefetch(bptr[0] + k0 + 64, 0, 3);
            __builtin_prefetch(bptr[2] + k0 + 64, 0, 3);
        }
        v8i a[2], b[4];
#pragma unroll
        for (int mi = 0; mi < 2; ++mi) {
            const int2* p = (const int2*)(aptr[mi] + k0);
            int2 d0 = p[0];
            int2 d1 = p[1];
            int2 d2 = p[4];
            int2 d3 = p[5];
            a[mi][0] = d0.x; a[mi][1] = d0.y;
            a[mi][2] = d1.x; a[mi][3] = d1.y;
            a[mi][4] = d2.x; a[mi][5] = d2.y;
            a[mi][6] = d3.x; a[mi][7] = d3.y;
        }
#pragma unroll
        for (int ni = 0; ni < 4; ++ni) {
            const int4* p = (const int4*)(bptr[ni] + k0);
            int4 e0 = p[0];
            int4 e1 = p[2];
            b[ni][0] = e0.x; b[ni][1] = e0.y; b[ni][2] = e0.z; b[ni][3] = e0.w;
            b[ni][4] = e1.x; b[ni][5] = e1.y; b[ni][6] = e1.z; b[ni][7] = e1.w;
        }
#pragma unroll
        for (int mi = 0; mi < 2; ++mi)
#pragma unroll
            for (int ni = 0; ni < 4; ++ni)
                acc[mi][ni] = __builtin_amdgcn_wmma_i32_16x16x64_iu8(
                     true, a[mi],  true, b[ni],
                    acc[mi][ni],  false,  false);
    }

#pragma unroll
    for (int ni = 0; ni < 4; ++ni) {
        int n = n_base + ni * 16 + l15;
        float sc = s[n];
        float bb = bq[n];
#pragma unroll
        for (int mi = 0; mi < 2; ++mi) {
            int m0 = m_base + mi * 16 + half * 8;
            float* o = out + (size_t)m0 * DOUT + n;
            v8i c = acc[mi][ni];
#pragma unroll
            for (int r = 0; r < 8; ++r)
                o[(size_t)r * DOUT] = (float)c[r] * sc + bb;
        }
    }
}

extern "C" void kernel_launch(void* const* d_in, const int* in_sizes, int n_in,
                              void* d_out, int out_size, void* d_ws, size_t ws_size,
                              hipStream_t stream) {
    const float* x    = (const float*)d_in[0];
    const float* w    = (const float*)d_in[1];
    const float* bias = (const float*)d_in[2];
    float* out = (float*)d_out;

    char* ws = (char*)d_ws;
    unsigned* mx = (unsigned*)ws;
    int8_t* xq = (int8_t*)(ws + OFF_XQ);
    int8_t* wq = (int8_t*)(ws + OFF_WQ);
    float*  sC = (float*)(ws + OFF_S);
    float*  bC = (float*)(ws + OFF_B);

    const int n4 = MTOT * DIN / 4;

    k_init<<<1, 1, 0, stream>>>(mx);
    k_absmax<<<2048, 256, 0, stream>>>((const float4*)x, mx, n4);
    k_quant_w<<<DOUT, 256, 0, stream>>>(w, bias, wq, sC, bC, mx);
    k_quant_x<<<4096, 256, 0, stream>>>((const float4*)x, (int*)xq, mx, n4);

    dim3 grid(MTOT / 128, DOUT / 128);
    k_gemm_iu8<<<grid, 256, 0, stream>>>(xq, wq, sC, bC, out);
}
